// GATEncoder_58085137711751
// MI455X (gfx1250) — compile-verified, matrix operand order checked by reading, not run on hardware
//
#include <hip/hip_runtime.h>
#include <math.h>

// ---------------------------------------------------------------------------
// GAT 2-layer pipeline for MI455X (gfx1250, wave32, WMMA)
//   L1: h1 = x(50000x128) @ W1(128x64)   -> v_wmma_f32_16x16x32_f16,
//       W1 staged once per block in LDS as pre-swizzled f16 fragments,
//       one wave computes a full 16x64 strip (A loaded/converted once).
//       3-pass segment softmax over 850k edges (8 heads), L2-resident atomics.
//   L2: h2 = elu_out(50000x64) @ W2(64x8) -> WMMA with zero-padded N tile.
// ---------------------------------------------------------------------------

typedef __attribute__((ext_vector_type(16))) _Float16 v16h;
typedef __attribute__((ext_vector_type(8)))  float    v8f;

#define NEG_SLOPE 0.2f
#define SM_EPS 1e-16f

// order-preserving float<->uint map for atomic max
__device__ __forceinline__ unsigned f2ord(float f) {
  unsigned u = __float_as_uint(f);
  return (u & 0x80000000u) ? ~u : (u | 0x80000000u);
}
__device__ __forceinline__ float ord2f(unsigned u) {
  unsigned i = (u & 0x80000000u) ? (u & 0x7fffffffu) : ~u;
  return __uint_as_float(i);
}
__device__ __forceinline__ void atomic_add_f32(float* p, float v) {
  __hip_atomic_fetch_add(p, v, __ATOMIC_RELAXED, __HIP_MEMORY_SCOPE_AGENT);
}
__device__ __forceinline__ void atomic_max_u32(unsigned* p, unsigned v) {
  __hip_atomic_fetch_max(p, v, __ATOMIC_RELAXED, __HIP_MEMORY_SCOPE_AGENT);
}
__device__ __forceinline__ float lrelu(float a) {
  return a > 0.f ? a : NEG_SLOPE * a;
}

// -------------------------- GEMM 1: x @ W1 (K=128, Ncols=64) ---------------
// Block = 256 threads = 8 waves. W1 -> LDS once as per-lane f16 B fragments.
// Each wave computes one 16x64 strip: A fragment loaded once per K-step,
// reused for 4 column tiles (4 WMMAs).
__global__ void __launch_bounds__(256) gemm1_wmma(const float* __restrict__ x,
                                                  const float* __restrict__ W,
                                                  float* __restrict__ h1, int N) {
  // bfrag[kt][tn][lane] : B fragment for K-step kt, column tile tn (16 KB LDS)
  __shared__ v16h bfrag[4][4][32];
  const int tid = threadIdx.x;
  for (int i = tid; i < 512; i += 256) {
    int kt = i >> 7, tn = (i >> 5) & 3, ln = i & 31;
    // ISA 7.12.2 16-bit B 32x16 layout (mirror of A): lane<16 -> K {0..7,16..23}
    int n  = tn * 16 + (ln & 15);
    int k0 = kt * 32 + ((ln < 16) ? 0 : 8);
    int k1 = k0 + 16;
    v16h b;
    #pragma unroll
    for (int e = 0; e < 8; ++e) b[e]     = (_Float16)W[(k0 + e) * 64 + n];
    #pragma unroll
    for (int e = 0; e < 8; ++e) b[8 + e] = (_Float16)W[(k1 + e) * 64 + n];
    bfrag[kt][tn][ln] = b;
  }
  __syncthreads();

  const int wave = tid >> 5, lane = tid & 31;
  const int tm = blockIdx.x * 8 + wave;
  if (tm >= (N >> 4)) return;                 // N == 50000 -> 3125 tiles
  const int m = tm * 16 + (lane & 15);
  const float* xr = x + (size_t)m * 128;

  v8f c0 = {}, c1 = {}, c2 = {}, c3 = {};
  #pragma unroll
  for (int kt = 0; kt < 4; ++kt) {
    int k0 = kt * 32 + ((lane < 16) ? 0 : 8);
    int k1 = k0 + 16;
    v16h a;
    // two contiguous 8-float runs -> vectorized b128 loads
    float4 r0 = *(const float4*)(xr + k0);
    float4 r1 = *(const float4*)(xr + k0 + 4);
    float4 r2 = *(const float4*)(xr + k1);
    float4 r3 = *(const float4*)(xr + k1 + 4);
    a[0] = (_Float16)r0.x; a[1] = (_Float16)r0.y; a[2]  = (_Float16)r0.z; a[3]  = (_Float16)r0.w;
    a[4] = (_Float16)r1.x; a[5] = (_Float16)r1.y; a[6]  = (_Float16)r1.z; a[7]  = (_Float16)r1.w;
    a[8] = (_Float16)r2.x; a[9] = (_Float16)r2.y; a[10] = (_Float16)r2.z; a[11] = (_Float16)r2.w;
    a[12] = (_Float16)r3.x; a[13] = (_Float16)r3.y; a[14] = (_Float16)r3.z; a[15] = (_Float16)r3.w;
    c0 = __builtin_amdgcn_wmma_f32_16x16x32_f16(false, a, false, bfrag[kt][0][lane], (short)0, c0, false, false);
    c1 = __builtin_amdgcn_wmma_f32_16x16x32_f16(false, a, false, bfrag[kt][1][lane], (short)0, c1, false, false);
    c2 = __builtin_amdgcn_wmma_f32_16x16x32_f16(false, a, false, bfrag[kt][2][lane], (short)0, c2, false, false);
    c3 = __builtin_amdgcn_wmma_f32_16x16x32_f16(false, a, false, bfrag[kt][3][lane], (short)0, c3, false, false);
  }
  // D layout: lane<16 -> M rows 0..7, lane>=16 -> rows 8..15; N = lane%16
  const int mb = tm * 16 + ((lane < 16) ? 0 : 8);
  const int nn = lane & 15;
  #pragma unroll
  for (int r = 0; r < 8; ++r) {
    float* row = h1 + (size_t)(mb + r) * 64 + nn;
    row[0] = c0[r]; row[16] = c1[r]; row[32] = c2[r]; row[48] = c3[r];
  }
}

// -------------------------- GEMM 2: h @ W2 (K=64, Ncols=8, zero-padded) ----
__global__ void gemm2_wmma(const float* __restrict__ h,
                           const float* __restrict__ W2,
                           float* __restrict__ h2, int N) {
  int wid  = (blockIdx.x * blockDim.x + threadIdx.x) >> 5;
  int lane = threadIdx.x & 31;
  if (wid >= (N >> 4)) return;
  int m  = wid * 16 + (lane & 15);
  int nl = lane & 15;
  v8f c = {};
  const float* hr = h + (size_t)m * 64;
  #pragma unroll
  for (int kt = 0; kt < 2; ++kt) {
    int k0 = kt * 32 + ((lane < 16) ? 0 : 8);
    int k1 = k0 + 16;
    v16h a, b;
    #pragma unroll
    for (int e = 0; e < 8; ++e) a[e]     = (_Float16)hr[k0 + e];
    #pragma unroll
    for (int e = 0; e < 8; ++e) a[8 + e] = (_Float16)hr[k1 + e];
    #pragma unroll
    for (int e = 0; e < 8; ++e) b[e]     = (nl < 8) ? (_Float16)W2[(k0 + e) * 8 + nl] : (_Float16)0.f;
    #pragma unroll
    for (int e = 0; e < 8; ++e) b[8 + e] = (nl < 8) ? (_Float16)W2[(k1 + e) * 8 + nl] : (_Float16)0.f;
    c = __builtin_amdgcn_wmma_f32_16x16x32_f16(false, a, false, b, (short)0, c, false, false);
  }
  if (nl < 8) {
    int mb = wid * 16 + ((lane < 16) ? 0 : 8);
    #pragma unroll
    for (int r = 0; r < 8; ++r) h2[(size_t)(mb + r) * 8 + nl] = c[r];
  }
}

// -------------------------- misc node kernels ------------------------------
__global__ void init1_k(float* __restrict__ out1, const float* __restrict__ b1,
                        unsigned* __restrict__ amax, float* __restrict__ denom,
                        int N) {
  int idx = blockIdx.x * blockDim.x + threadIdx.x;
  if (idx < N * 64) out1[idx] = b1[idx & 63];
  if (idx < N * 8) { amax[idx] = 0u; denom[idx] = 0.f; }
}

__global__ void scores1_k(const float* __restrict__ h1,
                          const float* __restrict__ att_s,
                          const float* __restrict__ att_d,
                          float* __restrict__ asrc, float* __restrict__ adst,
                          int N) {
  int idx = blockIdx.x * blockDim.x + threadIdx.x;
  if (idx >= N * 8) return;
  int n = idx >> 3, hd = idx & 7;
  const float* hp = h1 + (size_t)n * 64 + hd * 8;
  const float* as = att_s + hd * 8;
  const float* ad = att_d + hd * 8;
  float ss = 0.f, dd = 0.f;
  #pragma unroll
  for (int c = 0; c < 8; ++c) { float v = hp[c]; ss += v * as[c]; dd += v * ad[c]; }
  asrc[idx] = ss; adst[idx] = dd;
}

__global__ void elu_k(float* __restrict__ o, int total) {
  int idx = blockIdx.x * blockDim.x + threadIdx.x;
  if (idx >= total) return;
  float v = o[idx];
  o[idx] = v > 0.f ? v : (__expf(v) - 1.f);
}

__global__ void l2prep_k(const float* __restrict__ h2,
                         const float* __restrict__ att_s,
                         const float* __restrict__ att_d,
                         const float* __restrict__ b2,
                         float* __restrict__ out, float* __restrict__ asrc,
                         float* __restrict__ adst, unsigned* __restrict__ amax,
                         float* __restrict__ denom, int N) {
  int n = blockIdx.x * blockDim.x + threadIdx.x;
  if (n >= N) return;
  const float* hp = h2 + (size_t)n * 8;
  float ss = 0.f, dd = 0.f;
  #pragma unroll
  for (int c = 0; c < 8; ++c) {
    float v = hp[c]; ss += v * att_s[c]; dd += v * att_d[c];
    out[n * 8 + c] = b2[c];
  }
  asrc[n] = ss; adst[n] = dd; amax[n] = 0u; denom[n] = 0.f;
}

// -------------------------- edge passes, layer 1 (8 heads) -----------------
__device__ __forceinline__ void edge_sd(const int* ei, int e, int E, int& s, int& d) {
  if (e < E) { s = ei[e]; d = ei[E + e]; } else { s = d = e - E; }  // self loops
}

// thread per edge: float4 gathers of the 8-head score vectors
__global__ void edge_max1(const int* __restrict__ ei, const float* __restrict__ asrc,
                          const float* __restrict__ adst, unsigned* __restrict__ amax,
                          int E, int N) {
  int e = blockIdx.x * blockDim.x + threadIdx.x;
  if (e >= E + N) return;
  int s, d; edge_sd(ei, e, E, s, d);
  const float4* ap = (const float4*)(asrc + (size_t)s * 8);
  const float4* dp = (const float4*)(adst + (size_t)d * 8);
  float4 a0 = ap[0], a1 = ap[1], b0 = dp[0], b1 = dp[1];
  float v[8] = { a0.x + b0.x, a0.y + b0.y, a0.z + b0.z, a0.w + b0.w,
                 a1.x + b1.x, a1.y + b1.y, a1.z + b1.z, a1.w + b1.w };
  unsigned* am = amax + (size_t)d * 8;
  #pragma unroll
  for (int h = 0; h < 8; ++h) atomic_max_u32(am + h, f2ord(lrelu(v[h])));
}

__global__ void edge_sum1(const int* __restrict__ ei, const float* __restrict__ asrc,
                          const float* __restrict__ adst, const unsigned* __restrict__ amax,
                          float* __restrict__ alpha, float* __restrict__ denom,
                          int E, int N) {
  int e = blockIdx.x * blockDim.x + threadIdx.x;
  if (e >= E + N) return;
  int s, d; edge_sd(ei, e, E, s, d);
  const float4* ap = (const float4*)(asrc + (size_t)s * 8);
  const float4* dp = (const float4*)(adst + (size_t)d * 8);
  float4 a0 = ap[0], a1 = ap[1], b0 = dp[0], b1 = dp[1];
  float v[8] = { a0.x + b0.x, a0.y + b0.y, a0.z + b0.z, a0.w + b0.w,
                 a1.x + b1.x, a1.y + b1.y, a1.z + b1.z, a1.w + b1.w };
  const unsigned* am = amax + (size_t)d * 8;
  float* dn = denom + (size_t)d * 8;
  float w[8];
  #pragma unroll
  for (int h = 0; h < 8; ++h) {
    w[h] = __expf(lrelu(v[h]) - ord2f(am[h]));
    atomic_add_f32(dn + h, w[h]);
  }
  float4* alp = (float4*)(alpha + (size_t)e * 8);
  alp[0] = make_float4(w[0], w[1], w[2], w[3]);
  alp[1] = make_float4(w[4], w[5], w[6], w[7]);
}

// thread per (edge, head): 8 fp32 atomics each (accumulator is L2-resident)
__global__ void edge_agg1(const int* __restrict__ ei, const float* __restrict__ alpha,
                          const float* __restrict__ denom, const float* __restrict__ h1,
                          float* __restrict__ out1, int E, int N) {
  int idx = blockIdx.x * blockDim.x + threadIdx.x;
  if (idx >= (E + N) * 8) return;
  int e = idx >> 3, hd = idx & 7, s, d;
  edge_sd(ei, e, E, s, d);
  float w = alpha[idx] / (denom[(size_t)d * 8 + hd] + SM_EPS);
  const float4* hp = (const float4*)(h1 + (size_t)s * 64 + hd * 8);
  float4 p0 = hp[0], p1 = hp[1];
  float* o = out1 + (size_t)d * 64 + hd * 8;
  atomic_add_f32(o + 0, p0.x * w); atomic_add_f32(o + 1, p0.y * w);
  atomic_add_f32(o + 2, p0.z * w); atomic_add_f32(o + 3, p0.w * w);
  atomic_add_f32(o + 4, p1.x * w); atomic_add_f32(o + 5, p1.y * w);
  atomic_add_f32(o + 6, p1.z * w); atomic_add_f32(o + 7, p1.w * w);
}

// -------------------------- edge passes, layer 2 (1 head) ------------------
__global__ void edge_max2(const int* __restrict__ ei, const float* __restrict__ asrc,
                          const float* __restrict__ adst, unsigned* __restrict__ amax,
                          int E, int N) {
  int e = blockIdx.x * blockDim.x + threadIdx.x;
  if (e >= E + N) return;
  int s, d; edge_sd(ei, e, E, s, d);
  atomic_max_u32(&amax[d], f2ord(lrelu(asrc[s] + adst[d])));
}

__global__ void edge_sum2(const int* __restrict__ ei, const float* __restrict__ asrc,
                          const float* __restrict__ adst, const unsigned* __restrict__ amax,
                          float* __restrict__ alpha, float* __restrict__ denom,
                          int E, int N) {
  int e = blockIdx.x * blockDim.x + threadIdx.x;
  if (e >= E + N) return;
  int s, d; edge_sd(ei, e, E, s, d);
  float w = __expf(lrelu(asrc[s] + adst[d]) - ord2f(amax[d]));
  alpha[e] = w;
  atomic_add_f32(&denom[d], w);
}

__global__ void edge_agg2(const int* __restrict__ ei, const float* __restrict__ alpha,
                          const float* __restrict__ denom, const float* __restrict__ h2,
                          float* __restrict__ out, int E, int N) {
  int e = blockIdx.x * blockDim.x + threadIdx.x;
  if (e >= E + N) return;
  int s, d; edge_sd(ei, e, E, s, d);
  float w = alpha[e] / (denom[d] + SM_EPS);
  const float4* hp = (const float4*)(h2 + (size_t)s * 8);
  float4 p0 = hp[0], p1 = hp[1];
  float* o = out + (size_t)d * 8;
  atomic_add_f32(o + 0, p0.x * w); atomic_add_f32(o + 1, p0.y * w);
  atomic_add_f32(o + 2, p0.z * w); atomic_add_f32(o + 3, p0.w * w);
  atomic_add_f32(o + 4, p1.x * w); atomic_add_f32(o + 5, p1.y * w);
  atomic_add_f32(o + 6, p1.z * w); atomic_add_f32(o + 7, p1.w * w);
}

// ---------------------------------------------------------------------------
extern "C" void kernel_launch(void* const* d_in, const int* in_sizes, int n_in,
                              void* d_out, int out_size, void* d_ws, size_t ws_size,
                              hipStream_t stream) {
  const float* x        = (const float*)d_in[0];
  const int*   ei       = (const int*)d_in[1];     // [2,E]: src row, dst row
  const float* W1       = (const float*)d_in[2];
  const float* att_src1 = (const float*)d_in[3];
  const float* att_dst1 = (const float*)d_in[4];
  const float* b1       = (const float*)d_in[5];
  const float* W2       = (const float*)d_in[6];
  const float* att_src2 = (const float*)d_in[7];
  const float* att_dst2 = (const float*)d_in[8];
  const float* b2       = (const float*)d_in[9];
  float* out = (float*)d_out;

  const int N = in_sizes[0] / 128;   // 50000
  const int E = in_sizes[1] / 2;     // 800000
  const int Etot = E + N;

  float* ws = (float*)d_ws;
  float*    h1     = ws;                                   // N*64
  float*    out1   = h1 + (size_t)N * 64;                  // N*64
  float*    asrc1  = out1 + (size_t)N * 64;                // N*8
  float*    adst1  = asrc1 + (size_t)N * 8;                // N*8
  unsigned* amax1  = (unsigned*)(adst1 + (size_t)N * 8);   // N*8
  float*    denom1 = (float*)(amax1 + (size_t)N * 8);      // N*8
  float*    alpha  = denom1 + (size_t)N * 8;               // Etot*8 (reused for L2)
  float*    h2lin  = alpha + (size_t)Etot * 8;             // N*8
  float*    asrc2  = h2lin + (size_t)N * 8;                // N
  float*    adst2  = asrc2 + N;                            // N
  unsigned* amax2  = (unsigned*)(adst2 + N);               // N
  float*    denom2 = (float*)(amax2 + N);                  // N

  const int TPB = 256;
  auto blk = [](long long t, int tpb) { return (int)((t + tpb - 1) / tpb); };

  // ---- layer 1 ----
  init1_k<<<blk((long long)N * 64, TPB), TPB, 0, stream>>>(out1, b1, amax1, denom1, N);
  gemm1_wmma<<<blk((long long)(N / 16), 8), TPB, 0, stream>>>(x, W1, h1, N);
  scores1_k<<<blk((long long)N * 8, TPB), TPB, 0, stream>>>(h1, att_src1, att_dst1,
                                                            asrc1, adst1, N);
  edge_max1<<<blk(Etot, TPB), TPB, 0, stream>>>(ei, asrc1, adst1, amax1, E, N);
  edge_sum1<<<blk(Etot, TPB), TPB, 0, stream>>>(ei, asrc1, adst1, amax1, alpha, denom1, E, N);
  edge_agg1<<<blk((long long)Etot * 8, TPB), TPB, 0, stream>>>(ei, alpha, denom1, h1, out1, E, N);
  elu_k<<<blk((long long)N * 64, TPB), TPB, 0, stream>>>(out1, N * 64);

  // ---- layer 2 ----
  gemm2_wmma<<<blk((long long)(N / 16) * 32, TPB), TPB, 0, stream>>>(out1, W2, h2lin, N);
  l2prep_k<<<blk(N, TPB), TPB, 0, stream>>>(h2lin, att_src2, att_dst2, b2, out,
                                            asrc2, adst2, amax2, denom2, N);
  edge_max2<<<blk(Etot, TPB), TPB, 0, stream>>>(ei, asrc2, adst2, amax2, E, N);
  edge_sum2<<<blk(Etot, TPB), TPB, 0, stream>>>(ei, asrc2, adst2, amax2, alpha, denom2, E, N);
  edge_agg2<<<blk(Etot, TPB), TPB, 0, stream>>>(ei, alpha, denom2, h2lin, out, E, N);
}
